// CausalSelfAttention_2843268350635
// MI455X (gfx1250) — matrix operand order repaired and checked by reading, not compiled, not run on hardware
//
#include <hip/hip_runtime.h>

typedef unsigned int u32;
typedef unsigned short u16;

typedef __attribute__((ext_vector_type(16))) __bf16 v16bf;
typedef __attribute__((ext_vector_type(8)))  float  v8f;

union Frag {
  v16bf v;
  uint4 q[2];
  u32   u[8];
};

// ---------------- helpers ----------------

__device__ __forceinline__ u16 f2bf(float f) {
  u32 u = __float_as_uint(f);
  u += 0x7FFFu + ((u >> 16) & 1u);        // round-to-nearest-even
  return (u16)(u >> 16);
}
__device__ __forceinline__ u32 pk2(float a, float b) {
  return (u32)f2bf(a) | ((u32)f2bf(b) << 16);
}
__device__ __forceinline__ v8f wmma_bf16(const Frag& a, const Frag& b, v8f c) {
  return __builtin_amdgcn_wmma_f32_16x16x32_bf16(false, a.v, false, b.v,
                                                 (short)0, c, false, false);
}

// Problem constants
#define Bsz 4
#define Tsz 2048
#define Csz 1024
#define Hn  16
#define Dh  64

// =====================================================================
// Kernel 1: qkv = x @ Wqkv + bqkv, scattered to bf16 Q[B,H,T,D],
//           K[B,H,T,D], Vt[B,H,D,T].
// Block tile 128x128, k-step 32, 8 waves of 64x32.
// =====================================================================
__global__ __launch_bounds__(256) void qkv_gemm(const float* __restrict__ X,
                                                const float* __restrict__ W,
                                                const float* __restrict__ bias,
                                                u16* __restrict__ Qb,
                                                u16* __restrict__ Kb,
                                                u16* __restrict__ Vt) {
  __shared__ __attribute__((aligned(16))) u16 As[2][128 * 32];
  __shared__ __attribute__((aligned(16))) u16 Bs[2][128 * 32];

  const int t    = threadIdx.x;
  const int m0   = blockIdx.x * 128;
  const int n0   = blockIdx.y * 128;
  const int lane = t & 31;
  const int wave = t >> 5;
  const int wm   = (wave & 1) * 64;
  const int wn   = (wave >> 1) * 32;
  const int q15  = lane & 15;
  const int half = lane >> 4;

  v8f acc[4][2] = {};

  auto loadA = [&](int buf, int kk) {
    const int k0 = kk * 32;
#pragma unroll
    for (int f = 0; f < 4; ++f) {
      int fid = t + f * 256;
      int row = fid >> 3;
      int kq  = (fid & 7) * 4;
      float4 v = *(const float4*)(X + (size_t)(m0 + row) * 1024 + k0 + kq);
      uint2 uv; uv.x = pk2(v.x, v.y); uv.y = pk2(v.z, v.w);
      *(uint2*)&As[buf][row * 32 + kq] = uv;
    }
  };
  auto loadB = [&](int buf, int kk) {
    const int k0 = kk * 32;
#pragma unroll
    for (int f = 0; f < 4; ++f) {
      int fid  = t + f * 256;
      int krow = fid >> 5;
      int nq   = (fid & 31) * 4;
      float4 v = *(const float4*)(W + (size_t)(k0 + krow) * 3072 + n0 + nq);
      Bs[buf][(nq + 0) * 32 + krow] = f2bf(v.x);
      Bs[buf][(nq + 1) * 32 + krow] = f2bf(v.y);
      Bs[buf][(nq + 2) * 32 + krow] = f2bf(v.z);
      Bs[buf][(nq + 3) * 32 + krow] = f2bf(v.w);
    }
  };
  auto compute = [&](int buf) {
    Frag am[4], bn[2];
#pragma unroll
    for (int i = 0; i < 4; ++i) {
      int row = wm + i * 16 + q15;
      am[i].q[0] = *(const uint4*)&As[buf][row * 32 + half * 8];
      am[i].q[1] = *(const uint4*)&As[buf][row * 32 + 16 + half * 8];
    }
#pragma unroll
    for (int j = 0; j < 2; ++j) {
      int col = wn + j * 16 + q15;
      bn[j].q[0] = *(const uint4*)&Bs[buf][col * 32 + half * 8];
      bn[j].q[1] = *(const uint4*)&Bs[buf][col * 32 + 16 + half * 8];
    }
#pragma unroll
    for (int i = 0; i < 4; ++i)
#pragma unroll
      for (int j = 0; j < 2; ++j) acc[i][j] = wmma_bf16(am[i], bn[j], acc[i][j]);
  };

  loadA(0, 0); loadB(0, 0);
  __syncthreads();
  for (int kk = 0; kk < 32; ++kk) {
    int buf = kk & 1;
    if (kk + 1 < 32) { loadA(buf ^ 1, kk + 1); loadB(buf ^ 1, kk + 1); }
    compute(buf);
    __syncthreads();
  }

  // Epilogue: add bias, scatter bf16 to Q / K / Vt
#pragma unroll
  for (int i = 0; i < 4; ++i)
#pragma unroll
    for (int j = 0; j < 2; ++j) {
      int n = n0 + wn + j * 16 + q15;
      float bv = bias[n];
#pragma unroll
      for (int r = 0; r < 8; ++r) {
        int m  = m0 + wm + i * 16 + r + 8 * half;
        u16 o  = f2bf(acc[i][j][r] + bv);
        int bb = m >> 11;
        int tt = m & 2047;
        if (n < 1024) {
          int hh = n >> 6, d = n & 63;
          Qb[(((size_t)bb * Hn + hh) * Tsz + tt) * Dh + d] = o;
        } else if (n < 2048) {
          int n2 = n - 1024, hh = n2 >> 6, d = n2 & 63;
          Kb[(((size_t)bb * Hn + hh) * Tsz + tt) * Dh + d] = o;
        } else {
          int n2 = n - 2048, hh = n2 >> 6, d = n2 & 63;
          Vt[(((size_t)bb * Hn + hh) * Dh + d) * Tsz + tt] = o;
        }
      }
    }
}

// =====================================================================
// Kernel 2: flash attention, one wave per 16-query tile.
// S^T = K@Q^T (softmax over VGPR axis), O^T = V^T @ P^T.
// =====================================================================
__global__ __launch_bounds__(256) void attn_kernel(const u16* __restrict__ Qb,
                                                   const u16* __restrict__ Kb,
                                                   const u16* __restrict__ Vt,
                                                   u16* __restrict__ Y) {
  const int wave = threadIdx.x >> 5;
  const int lane = threadIdx.x & 31;
  const int q15  = lane & 15;
  const int half = lane >> 4;
  const int bh   = blockIdx.x >> 4;        // 0..63
  const int qg   = blockIdx.x & 15;
  const int qt   = qg * 8 + wave;          // 0..127
  const int b    = bh >> 4, h = bh & 15;
  const int q0   = qt * 16;

  const u16* Qrow  = Qb + ((size_t)bh * Tsz + q0 + q15) * Dh;
  const u16* Kbase = Kb + (size_t)bh * Tsz * Dh;
  const u16* Vbase = Vt + (size_t)bh * Dh * Tsz;

  // Q B-fragments (reused over the whole key loop)
  Frag qf[2];
#pragma unroll
  for (int s = 0; s < 2; ++s) {
    const uint4* p = (const uint4*)(Qrow + s * 32 + half * 8);
    qf[s].q[0] = p[0];
    qf[s].q[1] = p[2];
  }

  float m_run = -3.0e38f, l_run = 0.0f;
  v8f o[4] = {};

  const float scale = 0.125f;              // 1/sqrt(64)
  const int qglob = q0 + q15;
  const int nkb = (q0 + 47) >> 5;          // causal key-block count

  for (int kb = 0; kb < nkb; ++kb) {
    const int k0 = kb * 32;

    // S^T tiles: c[t2] = K_tile(16 keys x 64) @ Q^T
    v8f c[2] = {};
#pragma unroll
    for (int t2 = 0; t2 < 2; ++t2) {
      const u16* Krow = Kbase + (size_t)(k0 + t2 * 16 + q15) * Dh;
#pragma unroll
      for (int s = 0; s < 2; ++s) {
        Frag a;
        a.q[0] = *(const uint4*)(Krow + s * 32 + half * 8);
        a.q[1] = *(const uint4*)(Krow + s * 32 + 16 + half * 8);
        c[t2] = wmma_bf16(a, qf[s], c[t2]);
      }
    }

    // scale + causal mask + running softmax (keys live on VGPR axis)
    float p0[8], p1[8];
    float vmax = -3.0e38f;
#pragma unroll
    for (int r = 0; r < 8; ++r) {
      int key0 = k0 + r + 8 * half;
      float s0 = c[0][r] * scale; if (key0 > qglob)      s0 = -3.0e38f;
      float s1 = c[1][r] * scale; if (key0 + 16 > qglob) s1 = -3.0e38f;
      p0[r] = s0; p1[r] = s1;
      vmax = fmaxf(vmax, fmaxf(s0, s1));
    }
    vmax = fmaxf(vmax, __shfl_xor(vmax, 16, 32));
    float m_new = fmaxf(m_run, vmax);
    float corr  = __expf(m_run - m_new);
    m_run = m_new;
    float ssum = 0.0f;
#pragma unroll
    for (int r = 0; r < 8; ++r) {
      p0[r] = __expf(p0[r] - m_new);
      p1[r] = __expf(p1[r] - m_new);
      ssum += p0[r] + p1[r];
    }
    ssum += __shfl_xor(ssum, 16, 32);
    l_run = l_run * corr + ssum;
#pragma unroll
    for (int dt = 0; dt < 4; ++dt)
#pragma unroll
      for (int r = 0; r < 8; ++r) o[dt][r] *= corr;

    // repack exp'd S^T into B-fragment layout for P^T (lane = query)
    float oth0[8], oth1[8];
#pragma unroll
    for (int r = 0; r < 8; ++r) {
      oth0[r] = __shfl_xor(p0[r], 16, 32);
      oth1[r] = __shfl_xor(p1[r], 16, 32);
    }
    Frag pf;
#pragma unroll
    for (int j = 0; j < 4; ++j) {
      float a0 = half ? oth1[2 * j]     : p0[2 * j];
      float a1 = half ? oth1[2 * j + 1] : p0[2 * j + 1];
      float b0 = half ? p1[2 * j]       : oth0[2 * j];
      float b1 = half ? p1[2 * j + 1]   : oth0[2 * j + 1];
      pf.u[j]     = pk2(a0, a1);
      pf.u[j + 4] = pk2(b0, b1);
    }

    // O^T += V^T_tile @ P^T  (4 d-tiles of 16)
#pragma unroll
    for (int dt = 0; dt < 4; ++dt) {
      const u16* Vrow = Vbase + (size_t)(dt * 16 + q15) * Tsz;
      Frag av;
      av.q[0] = *(const uint4*)(Vrow + k0 + half * 16);
      av.q[1] = *(const uint4*)(Vrow + k0 + half * 16 + 8);
      o[dt] = wmma_bf16(av, pf, o[dt]);
    }
  }

  // normalize and write Y[b, t, h*64+d] as bf16
  const float inv = 1.0f / l_run;
  u16* Yrow = Y + ((size_t)(b * Tsz + q0 + q15)) * Csz + h * Dh;
#pragma unroll
  for (int dt = 0; dt < 4; ++dt)
#pragma unroll
    for (int r = 0; r < 8; ++r) {
      int d = dt * 16 + r + 8 * half;
      Yrow[d] = f2bf(o[dt][r] * inv);
    }
}

// =====================================================================
// Kernel 3: out = Y @ Wout + bout  (Y is bf16 in ws, out fp32)
// =====================================================================
__global__ __launch_bounds__(256) void proj_gemm(const u16* __restrict__ Yin,
                                                 const float* __restrict__ W,
                                                 const float* __restrict__ bias,
                                                 float* __restrict__ Out) {
  __shared__ __attribute__((aligned(16))) u16 As[2][128 * 32];
  __shared__ __attribute__((aligned(16))) u16 Bs[2][128 * 32];

  const int t    = threadIdx.x;
  const int m0   = blockIdx.x * 128;
  const int n0   = blockIdx.y * 128;
  const int lane = t & 31;
  const int wave = t >> 5;
  const int wm   = (wave & 1) * 64;
  const int wn   = (wave >> 1) * 32;
  const int q15  = lane & 15;
  const int half = lane >> 4;

  v8f acc[4][2] = {};

  auto loadA = [&](int buf, int kk) {
    const int k0 = kk * 32;
#pragma unroll
    for (int f = 0; f < 2; ++f) {
      int fid = t + f * 256;
      int row = fid >> 2;
      int kq  = (fid & 3) * 8;
      uint4 v = *(const uint4*)(Yin + (size_t)(m0 + row) * 1024 + k0 + kq);
      *(uint4*)&As[buf][row * 32 + kq] = v;
    }
  };
  auto loadB = [&](int buf, int kk) {
    const int k0 = kk * 32;
#pragma unroll
    for (int f = 0; f < 4; ++f) {
      int fid  = t + f * 256;
      int krow = fid >> 5;
      int nq   = (fid & 31) * 4;
      float4 v = *(const float4*)(W + (size_t)(k0 + krow) * 1024 + n0 + nq);
      Bs[buf][(nq + 0) * 32 + krow] = f2bf(v.x);
      Bs[buf][(nq + 1) * 32 + krow] = f2bf(v.y);
      Bs[buf][(nq + 2) * 32 + krow] = f2bf(v.z);
      Bs[buf][(nq + 3) * 32 + krow] = f2bf(v.w);
    }
  };
  auto compute = [&](int buf) {
    Frag am[4], bn[2];
#pragma unroll
    for (int i = 0; i < 4; ++i) {
      int row = wm + i * 16 + q15;
      am[i].q[0] = *(const uint4*)&As[buf][row * 32 + half * 8];
      am[i].q[1] = *(const uint4*)&As[buf][row * 32 + 16 + half * 8];
    }
#pragma unroll
    for (int j = 0; j < 2; ++j) {
      int col = wn + j * 16 + q15;
      bn[j].q[0] = *(const uint4*)&Bs[buf][col * 32 + half * 8];
      bn[j].q[1] = *(const uint4*)&Bs[buf][col * 32 + 16 + half * 8];
    }
#pragma unroll
    for (int i = 0; i < 4; ++i)
#pragma unroll
      for (int j = 0; j < 2; ++j) acc[i][j] = wmma_bf16(am[i], bn[j], acc[i][j]);
  };

  loadA(0, 0); loadB(0, 0);
  __syncthreads();
  for (int kk = 0; kk < 32; ++kk) {
    int buf = kk & 1;
    if (kk + 1 < 32) { loadA(buf ^ 1, kk + 1); loadB(buf ^ 1, kk + 1); }
    compute(buf);
    __syncthreads();
  }

#pragma unroll
  for (int i = 0; i < 4; ++i)
#pragma unroll
    for (int j = 0; j < 2; ++j) {
      int n = n0 + wn + j * 16 + q15;
      float bv = bias[n];
#pragma unroll
      for (int r = 0; r < 8; ++r) {
        int m = m0 + wm + i * 16 + r + 8 * half;
        Out[(size_t)m * 1024 + n] = acc[i][j][r] + bv;
      }
    }
}

// =====================================================================
// Launch
// =====================================================================
extern "C" void kernel_launch(void* const* d_in, const int* in_sizes, int n_in,
                              void* d_out, int out_size, void* d_ws, size_t ws_size,
                              hipStream_t stream) {
  const float* x    = (const float*)d_in[0];
  const float* Wqkv = (const float*)d_in[1];
  const float* bqkv = (const float*)d_in[2];
  const float* Wout = (const float*)d_in[3];
  const float* bout = (const float*)d_in[4];
  float* out = (float*)d_out;

  const size_t NE = (size_t)Bsz * Hn * Tsz * Dh;   // 8,388,608 bf16 elements
  u16* Qb = (u16*)d_ws;
  u16* Kb = Qb + NE;
  u16* Vt = Kb + NE;
  u16* Yb = Vt + NE;

  qkv_gemm<<<dim3(64, 24), 256, 0, stream>>>(x, Wqkv, bqkv, Qb, Kb, Vt);
  attn_kernel<<<dim3(1024), 256, 0, stream>>>(Qb, Kb, Vt, Yb);
  proj_gemm<<<dim3(64, 8), 256, 0, stream>>>(Yb, Wout, bout, out);
}
